// MultiTypeParamHeads_83966610637473
// MI455X (gfx1250) — matrix operand order repaired and checked by reading, not compiled, not run on hardware
//
#include <hip/hip_runtime.h>
#include <hip/hip_bf16.h>
#include <math.h>

typedef unsigned short u16;
typedef unsigned int   u32;
typedef __attribute__((ext_vector_type(16))) __bf16 bf16x16;
typedef __attribute__((ext_vector_type(8)))  float  f32x8;

#define BATCH      32768
#define IN_DIM     1024
#define HID        2048
#define HHID       1024
#define MAXP       12
#define NTYPES     8
#define TM         64          // rows per workgroup tile
#define H1_STRIDE  2056        // 2048 + 8 bf16 pad (anti bank conflict)

// force a batch of fragments to be materialized (batches DS loads -> single wait)
#define KEEP4(b0,b1,b2,b3) asm volatile("" :: "v"(b0), "v"(b1), "v"(b2), "v"(b3))

__constant__ int PARAM_COUNTS_D[NTYPES] = {3, 4, 6, 7, 9, 10, 12, 5};

__device__ __forceinline__ u16 f2bf(float x) {
    __bf16 h = (__bf16)x;                  // hardware v_cvt f32->bf16 (RNE)
    return __builtin_bit_cast(u16, h);
}

// erf via Abramowitz-Stegun 7.1.26 (|err| < 1.5e-7), v_rcp + v_exp transcendentals
__device__ __forceinline__ float gelu_fast(float x) {
    const float s  = x * 0.70710678118654752440f;
    const float az = fabsf(s);
    const float t  = __builtin_amdgcn_rcpf(1.0f + 0.3275911f * az);
    float y = t * (0.254829592f +
              t * (-0.284496736f +
              t * (1.421413741f +
              t * (-1.453152027f +
              t * 1.061405429f))));
    const float e = __expf(-az * az);
    float erfv = 1.0f - y * e;
    erfv = (s < 0.0f) ? -erfv : erfv;
    return 0.5f * x * (1.0f + erfv);
}

__device__ __forceinline__ f32x8 wmma_bf16(bf16x16 a, bf16x16 b, f32x8 c) {
    return __builtin_amdgcn_wmma_f32_16x16x32_bf16(
        false, a, false, b, (short)0, c, false, false);
}

// A fragment (16x32 bf16): lane l holds row rowBase + (l&15); K split in halves of 8
__device__ __forceinline__ bf16x16 load_frag_A(const u16* lds, int stride,
                                               int rowBase, int kBase, int lane) {
    const int m    = rowBase + (lane & 15);
    const int half = lane >> 4;
    const u16* p = lds + m * stride + kBase + half * 8;
    union { bf16x16 v; uint4 q[2]; } f;
    f.q[0] = *(const uint4*)(p);
    f.q[1] = *(const uint4*)(p + 16);
    return f.v;
}

// B fragment (32x16 bf16) from LDS stored TRANSPOSED [n][k]
__device__ __forceinline__ bf16x16 load_frag_B(const u16* lds, int stride,
                                               int nBase, int kBase, int lane) {
    const int n    = nBase + (lane & 15);
    const int half = lane >> 4;
    const u16* p = lds + n * stride + kBase + half * 8;
    union { bf16x16 v; uint4 q[2]; } f;
    f.q[0] = *(const uint4*)(p);
    f.q[1] = *(const uint4*)(p + 16);
    return f.v;
}

// ---------------- compaction / conversion kernels ----------------
__global__ void k_init(int* counts) {
    if (threadIdx.x < NTYPES) counts[threadIdx.x] = 0;
}
__global__ void k_count(const int* __restrict__ types, int* counts, int* rank) {
    int i = blockIdx.x * 256 + threadIdx.x;
    if (i < BATCH) rank[i] = atomicAdd(&counts[types[i]], 1);
}
__global__ void k_scan(const int* __restrict__ counts, int* basep) {
    if (threadIdx.x == 0) {
        int s = 0;
        for (int t = 0; t < NTYPES; ++t) { basep[t] = s; s += counts[t]; }
    }
}
__global__ void k_scatter(const int* __restrict__ types, const int* __restrict__ rank,
                          const int* __restrict__ basep, int* __restrict__ perm) {
    int i = blockIdx.x * 256 + threadIdx.x;
    if (i < BATCH) perm[basep[types[i]] + rank[i]] = i;
}
__global__ void k_cvt_bf16(const float* __restrict__ src, u16* __restrict__ dst, int n) {
    int i = (blockIdx.x * 256 + threadIdx.x) * 8;
    if (i >= n) return;
    float4 a = *(const float4*)(src + i);
    float4 b = *(const float4*)(src + i + 4);
    union { u16 s[8]; uint4 q; } o;
    o.s[0] = f2bf(a.x); o.s[1] = f2bf(a.y); o.s[2] = f2bf(a.z); o.s[3] = f2bf(a.w);
    o.s[4] = f2bf(b.x); o.s[5] = f2bf(b.y); o.s[6] = f2bf(b.z); o.s[7] = f2bf(b.w);
    *(uint4*)(dst + i) = o.q;
}

// ---------------- fused 3-layer head kernel ----------------
// grid = (maxTiles=512, NTYPES), block = 256 threads (8 wave32)
__global__ __launch_bounds__(256)
void k_fused(const float* __restrict__ x,
             const float* __restrict__ b1g,
             const float* __restrict__ b2g,
             const float* __restrict__ W3g,
             const float* __restrict__ b3g,
             const u16*   __restrict__ W1bf,
             const u16*   __restrict__ W2bf,
             const int*   __restrict__ perm,
             const int*   __restrict__ counts,
             const int*   __restrict__ basep,
             float* __restrict__ outP,
             float* __restrict__ outM) {
    __shared__ u16 h1s[TM * H1_STRIDE];                 // 257 KB: full h1 tile, bf16
    __shared__ union {
        struct { u16 xstg[TM * 72];  u16 w1stg[256 * 72]; } p1;          // 45 KB
        struct { u16 w2stg[128 * 72]; u16 h2slab[TM * 136];
                 u16 w3stg[16 * 136]; float pred[TM * 16]; } p2;         // 43.3 KB
    } sm;
    __shared__ int rowIdxS[TM];

    const int t = blockIdx.y;
    const int cnt = counts[t];
    const int tileBase = blockIdx.x * TM;
    if (tileBase >= cnt) return;
    const int bas   = basep[t];
    const int nrows = (cnt - tileBase < TM) ? (cnt - tileBase) : TM;

    const int tid  = threadIdx.x;
    const int lane = tid & 31;
    const int wave = tid >> 5;
    const int mw = wave >> 1;     // 0..3: 16-row block
    const int nw = wave & 1;      // 0..1: N half
    const int lh = lane >> 4;
    const int ln = lane & 15;

    if (tid < TM) rowIdxS[tid] = (tid < nrows) ? perm[bas + tileBase + tid] : 0;
    __syncthreads();

    // per-thread staging roles
    const int xm  = tid >> 2;           // X: row 0..63
    const int xkq = (tid & 3) << 4;     // X: k offset {0,16,32,48}
    const int wk4 = (tid & 15) << 2;    // W: k quad base 0..60
    const int wng = (tid >> 4) << 4;    // W1: n group base 0..240 (16 n each)
    const int wn8 = (tid >> 4) << 3;    // W2: n group base 0..120 (8 n each)

    // register prefetch state (software pipeline global->LDS)
    float4 xv[4];
    union RowW16 { uint4 q[2]; u16 s[16]; } wr[4];   // W1: 4 k-rows x 16 n
    union RowW8  { uint4 q;    u16 s[8];  } w2r[4];  // W2: 4 k-rows x 8 n

    auto prefetchX = [&](int kb) {
        if (xm < nrows) {
            const float* xp = x + (size_t)rowIdxS[xm] * IN_DIM + kb + xkq;
            xv[0] = ((const float4*)xp)[0];
            xv[1] = ((const float4*)xp)[1];
            xv[2] = ((const float4*)xp)[2];
            xv[3] = ((const float4*)xp)[3];
        }
    };
    auto storeX = [&]() {
        uint4* dst = (uint4*)&sm.p1.xstg[xm * 72 + xkq];
        if (xm < nrows) {
            union { u16 s[16]; uint4 q[4]; } o;
            #pragma unroll
            for (int i = 0; i < 4; ++i) {
                o.s[i*4+0] = f2bf(xv[i].x); o.s[i*4+1] = f2bf(xv[i].y);
                o.s[i*4+2] = f2bf(xv[i].z); o.s[i*4+3] = f2bf(xv[i].w);
            }
            dst[0] = o.q[0]; dst[1] = o.q[1]; dst[2] = o.q[2]; dst[3] = o.q[3];
        } else {
            uint4 z = {0,0,0,0};
            dst[0] = z; dst[1] = z; dst[2] = z; dst[3] = z;
        }
    };
    auto prefetchW1 = [&](int kb, int nbase) {
        const u16* gp = W1bf + ((size_t)(t * IN_DIM + kb + wk4)) * HID + nbase + wng;
        #pragma unroll
        for (int r = 0; r < 4; ++r) {
            wr[r].q[0] = ((const uint4*)gp)[0];
            wr[r].q[1] = ((const uint4*)gp)[1];
            gp += HID;
        }
    };
    auto storeW1 = [&]() {   // transpose: 4 k-values per column -> one b64 store
        #pragma unroll
        for (int j = 0; j < 16; ++j) {
            u32 lo = (u32)wr[0].s[j] | ((u32)wr[1].s[j] << 16);
            u32 hi = (u32)wr[2].s[j] | ((u32)wr[3].s[j] << 16);
            uint2 v; v.x = lo; v.y = hi;
            *(uint2*)&sm.p1.w1stg[(wng + j) * 72 + wk4] = v;
        }
    };
    auto prefetchW2 = [&](int kb, int ns2) {
        const u16* gp = W2bf + ((size_t)(t * HID + kb + wk4)) * HHID + ns2 * 128 + wn8;
        #pragma unroll
        for (int r = 0; r < 4; ++r) { w2r[r].q = *(const uint4*)gp; gp += HHID; }
    };
    auto storeW2 = [&]() {
        #pragma unroll
        for (int j = 0; j < 8; ++j) {
            u32 lo = (u32)w2r[0].s[j] | ((u32)w2r[1].s[j] << 16);
            u32 hi = (u32)w2r[2].s[j] | ((u32)w2r[3].s[j] << 16);
            uint2 v; v.x = lo; v.y = hi;
            *(uint2*)&sm.p2.w2stg[(wn8 + j) * 72 + wk4] = v;
        }
    };

    // ================= Phase 1: h1 = gelu(x @ W1 + b1), kept in LDS =================
    prefetchX(0);
    prefetchW1(0, 0);
    for (int nsl = 0; nsl < 8; ++nsl) {                 // 8 x 256-col slabs of HID
        const int nbase = nsl * 256;
        f32x8 c[8];
        #pragma unroll
        for (int i = 0; i < 8; ++i) c[i] = (f32x8){0,0,0,0,0,0,0,0};

        for (int kb = 0; kb < IN_DIM; kb += 64) {
            __syncthreads();                 // LDS consumers of previous chunk done
            storeX();
            storeW1();
            // prefetch next chunk (overlaps with this chunk's WMMAs)
            int nkb = kb + 64, nnsl = nsl;
            if (nkb == IN_DIM) { nkb = 0; nnsl = nsl + 1; }
            if (nnsl < 8) { prefetchX(nkb); prefetchW1(nkb, nnsl * 256); }
            __syncthreads();
            #pragma unroll
            for (int ks = 0; ks < 2; ++ks) {
                bf16x16 a  = load_frag_A(sm.p1.xstg, 72, mw * 16, ks * 32, lane);
                bf16x16 b0 = load_frag_B(sm.p1.w1stg, 72, nw*128 +  0, ks*32, lane);
                bf16x16 b1 = load_frag_B(sm.p1.w1stg, 72, nw*128 + 16, ks*32, lane);
                bf16x16 b2 = load_frag_B(sm.p1.w1stg, 72, nw*128 + 32, ks*32, lane);
                bf16x16 b3 = load_frag_B(sm.p1.w1stg, 72, nw*128 + 48, ks*32, lane);
                KEEP4(b0, b1, b2, b3);
                c[0] = wmma_bf16(a, b0, c[0]);
                c[1] = wmma_bf16(a, b1, c[1]);
                c[2] = wmma_bf16(a, b2, c[2]);
                c[3] = wmma_bf16(a, b3, c[3]);
                bf16x16 b4 = load_frag_B(sm.p1.w1stg, 72, nw*128 +  64, ks*32, lane);
                bf16x16 b5 = load_frag_B(sm.p1.w1stg, 72, nw*128 +  80, ks*32, lane);
                bf16x16 b6 = load_frag_B(sm.p1.w1stg, 72, nw*128 +  96, ks*32, lane);
                bf16x16 b7 = load_frag_B(sm.p1.w1stg, 72, nw*128 + 112, ks*32, lane);
                KEEP4(b4, b5, b6, b7);
                c[4] = wmma_bf16(a, b4, c[4]);
                c[5] = wmma_bf16(a, b5, c[5]);
                c[6] = wmma_bf16(a, b6, c[6]);
                c[7] = wmma_bf16(a, b7, c[7]);
            }
        }
        // epilogue: bias + GELU + bf16, into resident h1 tile
        #pragma unroll
        for (int t8 = 0; t8 < 8; ++t8) {
            const int col = nbase + nw*128 + t8*16 + ln;
            const float bb = b1g[t * HID + col];
            #pragma unroll
            for (int r = 0; r < 8; ++r) {
                const int m = mw*16 + lh*8 + r;
                h1s[m * H1_STRIDE + col] = f2bf(gelu_fast(c[t8][r] + bb));
            }
        }
    }
    __syncthreads();

    // ====== Phase 2+3: h2 = gelu(h1 @ W2 + b2) per 128-col slab; fused pred += h2 @ W3 ======
    f32x8 predC = (f32x8){0,0,0,0,0,0,0,0};
    prefetchW2(0, 0);
    for (int ns2 = 0; ns2 < 8; ++ns2) {                 // 8 x 128-col slabs of HHID
        f32x8 c2[4];
        #pragma unroll
        for (int i = 0; i < 4; ++i) c2[i] = (f32x8){0,0,0,0,0,0,0,0};

        for (int kb = 0; kb < HID; kb += 64) {
            __syncthreads();
            storeW2();
            int nkb = kb + 64, nns2 = ns2;
            if (nkb == HID) { nkb = 0; nns2 = ns2 + 1; }
            if (nns2 < 8) prefetchW2(nkb, nns2);
            __syncthreads();
            #pragma unroll
            for (int ks = 0; ks < 2; ++ks) {
                bf16x16 a  = load_frag_A(h1s, H1_STRIDE, mw * 16, kb + ks * 32, lane);
                bf16x16 b0 = load_frag_B(sm.p2.w2stg, 72, nw*64 +  0, ks*32, lane);
                bf16x16 b1 = load_frag_B(sm.p2.w2stg, 72, nw*64 + 16, ks*32, lane);
                bf16x16 b2 = load_frag_B(sm.p2.w2stg, 72, nw*64 + 32, ks*32, lane);
                bf16x16 b3 = load_frag_B(sm.p2.w2stg, 72, nw*64 + 48, ks*32, lane);
                KEEP4(b0, b1, b2, b3);
                c2[0] = wmma_bf16(a, b0, c2[0]);
                c2[1] = wmma_bf16(a, b1, c2[1]);
                c2[2] = wmma_bf16(a, b2, c2[2]);
                c2[3] = wmma_bf16(a, b3, c2[3]);
            }
        }
        __syncthreads();
        // epilogue: h2 slab -> LDS bf16; also stage W3 chunk transposed [p][k]
        #pragma unroll
        for (int t4 = 0; t4 < 4; ++t4) {
            const int col = nw*64 + t4*16 + ln;
            const float bb = b2g[t * HHID + ns2 * 128 + col];
            #pragma unroll
            for (int r = 0; r < 8; ++r) {
                const int m = mw*16 + lh*8 + r;
                sm.p2.h2slab[m * 136 + col] = f2bf(gelu_fast(c2[t4][r] + bb));
            }
        }
        for (int idx = tid; idx < 16 * 128; idx += 256) {
            const int p = idx >> 7, k = idx & 127;
            u16 v = 0;
            if (p < MAXP)
                v = f2bf(W3g[((size_t)(t * HHID + ns2 * 128 + k)) * MAXP + p]);
            sm.p2.w3stg[p * 136 + k] = v;
        }
        __syncthreads();
        // GEMM3 partial: waves split the slab K across nw halves
        #pragma unroll
        for (int ks = 0; ks < 2; ++ks) {
            const int kk = nw * 64 + ks * 32;
            bf16x16 a = load_frag_A(sm.p2.h2slab, 136, mw * 16, kk, lane);
            bf16x16 b = load_frag_B(sm.p2.w3stg, 136, 0, kk, lane);
            predC = wmma_bf16(a, b, predC);
        }
        __syncthreads();
    }

    // cross-wave K reduction of pred, then masked scatter to output
    for (int idx = tid; idx < TM * 16; idx += 256) sm.p2.pred[idx] = 0.0f;
    __syncthreads();
    #pragma unroll
    for (int r = 0; r < 8; ++r) {
        const int m = mw*16 + lh*8 + r;
        atomicAdd(&sm.p2.pred[m * 16 + ln], predC[r]);
    }
    __syncthreads();
    const int pc = PARAM_COUNTS_D[t];
    for (int idx = tid; idx < TM * MAXP; idx += 256) {
        const int m = idx / MAXP, p = idx - m * MAXP;
        if (m < nrows) {
            const int row = rowIdxS[m];
            const float wm = (p < pc) ? 1.0f : 0.0f;
            outP[(size_t)row * MAXP + p] = (sm.p2.pred[m * 16 + p] + b3g[t * MAXP + p]) * wm;
            outM[(size_t)row * MAXP + p] = wm;
        }
    }
}

extern "C" void kernel_launch(void* const* d_in, const int* in_sizes, int n_in,
                              void* d_out, int out_size, void* d_ws, size_t ws_size,
                              hipStream_t stream) {
    const float* x   = (const float*)d_in[0];
    const float* W1  = (const float*)d_in[1];
    const float* b1  = (const float*)d_in[2];
    const float* W2  = (const float*)d_in[3];
    const float* b2  = (const float*)d_in[4];
    const float* W3  = (const float*)d_in[5];
    const float* b3  = (const float*)d_in[6];
    const int* types = (const int*)d_in[7];

    char* ws = (char*)d_ws;
    int* counts = (int*)(ws + 0);
    int* basep  = (int*)(ws + 64);
    int* rank   = (int*)(ws + 256);
    int* perm   = (int*)(ws + 256 + BATCH * 4);
    u16* W1bf   = (u16*)(ws + 262400);                         // 8*1024*2048 bf16
    u16* W2bf   = (u16*)(ws + 262400 + (size_t)33554432);      // 8*2048*1024 bf16

    float* outP = (float*)d_out;
    float* outM = outP + (size_t)BATCH * MAXP;

    k_init   <<<1,   32,  0, stream>>>(counts);
    k_count  <<<128, 256, 0, stream>>>(types, counts, rank);
    k_scan   <<<1,   1,   0, stream>>>(counts, basep);
    k_scatter<<<128, 256, 0, stream>>>(types, rank, basep, perm);
    k_cvt_bf16<<<NTYPES*IN_DIM*HID/(256*8), 256, 0, stream>>>(W1, W1bf, NTYPES*IN_DIM*HID);
    k_cvt_bf16<<<NTYPES*HID*HHID/(256*8), 256, 0, stream>>>(W2, W2bf, NTYPES*HID*HHID);

    dim3 grid(BATCH / TM, NTYPES);   // worst case: all rows one type
    k_fused<<<grid, 256, 0, stream>>>(x, b1, b2, W3, b3, W1bf, W2bf,
                                      perm, counts, basep, outP, outM);
}
